// VectorQuantizerEMA_45140106281590
// MI455X (gfx1250) — matrix operand order repaired and checked by reading, not compiled, not run on hardware
//
#include <hip/hip_runtime.h>
#include <hip/hip_bf16.h>

typedef __attribute__((ext_vector_type(16))) __bf16 v16bf;
typedef __attribute__((ext_vector_type(8)))  float  v8f;
typedef __attribute__((ext_vector_type(4)))  float  v4f;
typedef __attribute__((ext_vector_type(2)))  float  v2f;

#define NUM_EMB   512
#define EMB_DIM   64
#define N_ROWS    131072      // 32 * 64 * 64 flattened vectors
#define HW        4096        // 64*64
#define CH_STRIDE 4096        // channel stride in NCHW (floats)
#define B_STRIDE  262144      // batch stride in NCHW (floats)
#define ROWS_PER_BLOCK 32

// d_out layout (floats): [0]=loss, [1..8388608]=quantized NCHW,
// [8388609]=perplexity, [8388610..]=encodings [N,512]
#define OUT_QUANT_OFF 1
#define OUT_PERP_OFF  8388609
#define OUT_ENC_OFF   8388610

// d_ws layout (floats):
//   [0]           loss accumulator
//   [1..512]      counts (u32)
//   [513..1024]   e_norm[k]
//   [1032.. ]     bf16 B-fragment table, 2048 entries * 32B = 64KB
//                 (byte offset 4128, 32B aligned)
#define WS_BTAB_OFF 1032

// Pre-swizzle E into WMMA B-fragment layout (bf16 32x16 tiles):
// entry e = (T*2 + s)*32 + L holds lane L's v16bf for column-tile T, k-step s.
// Lane L: col = 16T + (L&15); K = 2v + (L<16?0:16) + 32s, pairs {K, K+1}.
__global__ void vq_prep(const float* __restrict__ emb, float* __restrict__ ws) {
    const int t = blockIdx.x * blockDim.x + threadIdx.x;   // 2048 threads
    if (t == 0) ws[0] = 0.0f;
    if (t < NUM_EMB) {
        ((unsigned int*)ws)[1 + t] = 0u;
        const float* e = emb + t * EMB_DIM;
        float s = 0.0f;
        #pragma unroll
        for (int d = 0; d < EMB_DIM; ++d) s += e[d] * e[d];
        ws[513 + t] = s;
    }
    {
        const int L    = t & 31;
        const int s    = (t >> 5) & 1;
        const int T    = t >> 6;
        const int col  = T * 16 + (L & 15);
        const int koff = ((L < 16) ? 0 : 16) + 32 * s;
        const float* e = emb + (size_t)col * EMB_DIM;
        v16bf frag;
        #pragma unroll
        for (int v = 0; v < 8; ++v) {
            frag[2*v]   = (__bf16)e[koff + 2*v];
            frag[2*v+1] = (__bf16)e[koff + 2*v + 1];
        }
        ((v16bf*)(ws + WS_BTAB_OFF))[t] = frag;
    }
}

__launch_bounds__(256)
__global__ void vq_main(const float* __restrict__ x,    // NCHW [32,64,64,64]
                        const float* __restrict__ emb,  // [512,64]
                        float* __restrict__ out,
                        float* __restrict__ ws) {
    __shared__ float s_bv[2][16][4];
    __shared__ int   s_bi[2][16][4];
    __shared__ int   s_idx[ROWS_PER_BLOCK];

    const int lane = threadIdx.x & 31;
    const int wave = threadIdx.x >> 5;
    const int row_tile  = wave >> 2;   // 0..1  (16 rows each)
    const int col_group = wave & 3;    // 0..3  (128 cols each)
    const int half = lane >> 4;        // wave32 half
    const int l16  = lane & 15;

    const int nb = blockIdx.x * ROWS_PER_BLOCK;
    const int n  = nb + row_tile * 16 + l16;   // this lane's A row
    const int bb = n >> 12;
    const int hw = n & (HW - 1);
    const float* xbase = x + (size_t)bb * B_STRIDE + hw;

    // ---- A fragments: two bf16 16x32 tiles (K = 0..31, 32..63) ----
    v16bf a0, a1;
    #pragma unroll
    for (int v = 0; v < 8; ++v) {
        const int K = 2 * v + (half ? 16 : 0);
        a0[2*v]   = (__bf16)xbase[(size_t)(K     ) * CH_STRIDE];
        a0[2*v+1] = (__bf16)xbase[(size_t)(K +  1) * CH_STRIDE];
        a1[2*v]   = (__bf16)xbase[(size_t)(K + 32) * CH_STRIDE];
        a1[2*v+1] = (__bf16)xbase[(size_t)(K + 33) * CH_STRIDE];
    }

    // ---- running argmin of (||e||^2 - 2 x.e) per accumulator row ----
    float best_v[8];
    int   best_i[8];
    #pragma unroll
    for (int r = 0; r < 8; ++r) { best_v[r] = 3.4e38f; best_i[r] = 0; }

    const v16bf* btab = (const v16bf*)(ws + WS_BTAB_OFF);

    #pragma unroll 4
    for (int t = 0; t < 8; ++t) {
        const int T   = col_group * 8 + t;          // global column tile
        const int col = T * 16 + l16;               // this lane's B column
        const v16bf b0 = btab[(T * 2 + 0) * 32 + lane];
        const v16bf b1 = btab[(T * 2 + 1) * 32 + lane];
        v8f acc = {};
        acc = __builtin_amdgcn_wmma_f32_16x16x32_bf16(false, a0, false, b0,
                                                      (short)0, acc, false, false);
        acc = __builtin_amdgcn_wmma_f32_16x16x32_bf16(false, a1, false, b1,
                                                      (short)0, acc, false, false);
        const float en = ws[513 + col];
        #pragma unroll
        for (int r = 0; r < 8; ++r) {
            const float dist = en - 2.0f * acc[r];   // row M = r + half*8
            if (dist < best_v[r] || (dist == best_v[r] && col < best_i[r])) {
                best_v[r] = dist; best_i[r] = col;
            }
        }
    }

    // reduce across the 16 lanes of each half (lanes differ in bits 0..3)
    #pragma unroll
    for (int off = 8; off >= 1; off >>= 1) {
        #pragma unroll
        for (int r = 0; r < 8; ++r) {
            const float ov = __shfl_xor(best_v[r], off, 32);
            const int   oi = __shfl_xor(best_i[r], off, 32);
            if (ov < best_v[r] || (ov == best_v[r] && oi < best_i[r])) {
                best_v[r] = ov; best_i[r] = oi;
            }
        }
    }
    if (l16 == 0) {
        #pragma unroll
        for (int r = 0; r < 8; ++r) {
            s_bv[row_tile][half * 8 + r][col_group] = best_v[r];
            s_bi[row_tile][half * 8 + r][col_group] = best_i[r];
        }
    }
    __syncthreads();

    // finalize argmin across the 4 column groups: one row per thread 0..31
    if (threadIdx.x < ROWS_PER_BLOCK) {
        const int rt = threadIdx.x >> 4, rm = threadIdx.x & 15;
        float bv = s_bv[rt][rm][0];
        int   bi = s_bi[rt][rm][0];
        #pragma unroll
        for (int g = 1; g < 4; ++g) {
            const float v = s_bv[rt][rm][g];
            const int   i = s_bi[rt][rm][g];
            if (v < bv || (v == bv && i < bi)) { bv = v; bi = i; }
        }
        s_idx[threadIdx.x] = bi;
    }
    __syncthreads();

    // ---- encodings: one-hot rows, non-temporal stores ----
    // Row base byte address is 8 mod 16, so cols 2..509 form 127 aligned
    // float4 chunks; cols {0,1} and {510,511} are 8B-aligned float2 edges.
    float* out_enc = out + OUT_ENC_OFF;
    for (int j = threadIdx.x; j < ROWS_PER_BLOCK * 128; j += 256) {
        const int row = j >> 7;
        const int q   = j & 127;
        const int idx = s_idx[row];
        float* rowp = out_enc + (size_t)(nb + row) * NUM_EMB;
        if (q == 127) {
            v2f h, tl;
            h.x  = (idx == 0)   ? 1.0f : 0.0f;
            h.y  = (idx == 1)   ? 1.0f : 0.0f;
            tl.x = (idx == 510) ? 1.0f : 0.0f;
            tl.y = (idx == 511) ? 1.0f : 0.0f;
            __builtin_nontemporal_store(h,  (v2f*)rowp);
            __builtin_nontemporal_store(tl, (v2f*)(rowp + 510));
        } else {
            const int cbase = 2 + 4 * q;
            v4f val = {0.0f, 0.0f, 0.0f, 0.0f};
            const unsigned rel = (unsigned)(idx - cbase);
            if (rel < 4u) val[rel] = 1.0f;
            __builtin_nontemporal_store(val, (v4f*)(rowp + cbase));
        }
    }

    // ---- quantized (NCHW) + commitment-loss partial ----
    float lsum = 0.0f;
    for (int j = threadIdx.x; j < ROWS_PER_BLOCK * EMB_DIM; j += 256) {
        const int row = j >> 6;
        const int c   = j & 63;
        const int nn  = nb + row;
        const size_t off = (size_t)(nn >> 12) * B_STRIDE +
                           (size_t)c * CH_STRIDE + (nn & (HW - 1));
        const float q  = emb[(size_t)s_idx[row] * EMB_DIM + c];
        const float xv = x[off];
        __builtin_nontemporal_store(q, out + OUT_QUANT_OFF + off);
        const float d = q - xv;
        lsum += d * d;
    }
    #pragma unroll
    for (int off = 16; off >= 1; off >>= 1) lsum += __shfl_xor(lsum, off, 32);
    if (lane == 0) atomicAdd(&ws[0], lsum);

    // ---- index histogram for perplexity ----
    if (threadIdx.x < ROWS_PER_BLOCK)
        atomicAdd(((unsigned int*)ws) + 1 + s_idx[threadIdx.x], 1u);
}

__global__ void vq_final(float* __restrict__ out, const float* __restrict__ ws) {
    __shared__ float sh[NUM_EMB];
    const int t = threadIdx.x;
    const unsigned cnt = ((const unsigned int*)ws)[1 + t];
    const float p = (float)cnt / (float)N_ROWS;
    sh[t] = p * logf(p + 1e-10f);
    __syncthreads();
    for (int s = NUM_EMB / 2; s >= 1; s >>= 1) {
        if (t < s) sh[t] += sh[t + s];
        __syncthreads();
    }
    if (t == 0) {
        out[OUT_PERP_OFF] = expf(-sh[0]);
        out[0] = 0.25f * ws[0] / (float)((size_t)N_ROWS * EMB_DIM);
    }
}

extern "C" void kernel_launch(void* const* d_in, const int* in_sizes, int n_in,
                              void* d_out, int out_size, void* d_ws, size_t ws_size,
                              hipStream_t stream) {
    const float* x   = (const float*)d_in[0];   // [32,64,64,64] f32 NCHW
    const float* emb = (const float*)d_in[1];   // [512,64] f32
    float* out = (float*)d_out;
    float* ws  = (float*)d_ws;

    vq_prep<<<8, 256, 0, stream>>>(emb, ws);
    vq_main<<<N_ROWS / ROWS_PER_BLOCK, 256, 0, stream>>>(x, emb, out, ws);
    vq_final<<<1, NUM_EMB, 0, stream>>>(out, ws);
}
